// Comp_Attn_71322226917512
// MI455X (gfx1250) — matrix operand order repaired and checked by reading, not compiled, not run on hardware
//
#include <hip/hip_runtime.h>
#include <hip/hip_fp16.h>

typedef __attribute__((ext_vector_type(16))) _Float16 v16h;
typedef __attribute__((ext_vector_type(8)))  float    v8f;
typedef __attribute__((ext_vector_type(4)))  float    f32x4;

#define B_   4
#define C_   256
#define N_   4096
#define NB_  256          // N/16 tiles
#define LDSTRIDE 4100     // 4096 + 4 pad (bank spread)

static __device__ inline v8f wmma_f16(v16h a, v16h b, v8f c) {
  return __builtin_amdgcn_wmma_f32_16x16x32_f16(false, a, false, b, (short)0, c, false, false);
}

// ---------------------------------------------------------------------------
// Kernel 0: fold c1..c4 into q/k weights: Wqc[64,256], Wkc[64,256] (+ biases)
// ---------------------------------------------------------------------------
__global__ void combine_weights(const float* __restrict__ qw, const float* __restrict__ qb,
                                const float* __restrict__ kw, const float* __restrict__ kb,
                                const float* __restrict__ c1w, const float* __restrict__ c1b,
                                const float* __restrict__ c2w, const float* __restrict__ c2b,
                                const float* __restrict__ c3w, const float* __restrict__ c3b,
                                const float* __restrict__ c4w, const float* __restrict__ c4b,
                                float* __restrict__ qcW, float* __restrict__ qcB,
                                float* __restrict__ kcW, float* __restrict__ kcB) {
  int t = threadIdx.x;  // 256 threads, 1 block
  for (int o = t; o < 64 * 256; o += 256) {
    int row = o >> 8, c = o & 255;
    const float* cwq = (row < 32) ? (c1w + row * 32) : (c2w + (row - 32) * 32);
    const float* cwk = (row < 32) ? (c3w + row * 32) : (c4w + (row - 32) * 32);
    float sq = 0.f, sk = 0.f;
    for (int m = 0; m < 32; ++m) {
      sq += cwq[m] * qw[m * 256 + c];
      sk += cwk[m] * kw[m * 256 + c];
    }
    qcW[o] = sq; kcW[o] = sk;
  }
  if (t < 64) {
    int row = t;
    const float* cwq = (row < 32) ? (c1w + row * 32) : (c2w + (row - 32) * 32);
    const float* cwk = (row < 32) ? (c3w + row * 32) : (c4w + (row - 32) * 32);
    float sq = (row < 32) ? c1b[row] : c2b[row - 32];
    float sk = (row < 32) ? c3b[row] : c4b[row - 32];
    for (int m = 0; m < 32; ++m) { sq += cwq[m] * qb[m]; sk += cwk[m] * kb[m]; }
    qcB[row] = sq; kcB[row] = sk;
  }
}

// ---------------------------------------------------------------------------
// Kernel A: projections -> f16 WMMA fragments (pre-swizzled layouts)
//   qAf: [b][rowblk][kchunk][lane][16]  (A fragment, 16x32 f16 layout)
//   kBf: [b][colblk][kchunk][lane][16]  (B fragment, 32x16 f16 layout)
//   vBf: [b][cblk][jchunk][lane][16]    (B fragment of v^T for out-bmm)
// ---------------------------------------------------------------------------
__global__ void proj_scatter(const float* __restrict__ x,
                             const float* __restrict__ qcW, const float* __restrict__ qcB,
                             const float* __restrict__ kcW, const float* __restrict__ kcB,
                             const float* __restrict__ vw,  const float* __restrict__ vb,
                             _Float16* __restrict__ qAf, _Float16* __restrict__ kBf,
                             _Float16* __restrict__ vBf) {
  __shared__ float xs[256 * 64];  // 64 KB tile: [c][pp]
  int b = blockIdx.y;
  int pbase = blockIdx.x * 64;
  int t = threadIdx.x;
  const float* xb = x + (size_t)b * C_ * N_;
  for (int idx = t; idx < 256 * 64; idx += 256) {
    int c = idx >> 6, pp = idx & 63;
    xs[idx] = xb[(size_t)c * N_ + pbase + pp];
  }
  __syncthreads();

  int pp = t & 63;
  int rgrp = t >> 6;  // 0..3 -> rows [96*rgrp, 96*rgrp+96); uniform per wave
  int p = pbase + pp;
  for (int rr = 0; rr < 96; ++rr) {
    int row = rgrp * 96 + rr;
    const float* w; float bias; int kind, lrow;
    if (row < 64)       { lrow = row;       w = qcW + lrow * 256; bias = qcB[lrow]; kind = 0; }
    else if (row < 128) { lrow = row - 64;  w = kcW + lrow * 256; bias = kcB[lrow]; kind = 1; }
    else                { lrow = row - 128; w = vw  + lrow * 256; bias = vb[lrow];  kind = 2; }
    float s = bias;
    #pragma unroll 8
    for (int c = 0; c < 256; ++c) s += w[c] * xs[c * 64 + pp];
    _Float16 hv = (_Float16)s;
    if (kind == 0) {        // A-fragment lane map (16x32 f16)
      int k32 = lrow & 31, kch = lrow >> 5;
      int g = (k32 >> 3) & 1, h = (k32 & 7) + 8 * (k32 >> 4);
      int lane = (p & 15) + 16 * g;
      size_t idx = (((size_t)(b * NB_ + (p >> 4))) * 2 + kch) * 512 + lane * 16 + h;
      qAf[idx] = hv;
    } else if (kind == 1) { // B-fragment lane map (32x16 f16)
      int k32 = lrow & 31, kch = lrow >> 5;
      int g = (k32 >> 3) & 1, h = (k32 & 7) + 8 * (k32 >> 4);
      int lane = (p & 15) + 16 * g;
      size_t idx = (((size_t)(b * NB_ + (p >> 4))) * 2 + kch) * 512 + lane * 16 + h;
      kBf[idx] = hv;
    } else {                // v^T B-fragment: n = c%16, K = j%32
      int cblk = lrow >> 4, n = lrow & 15;
      int p32 = p & 31;
      int g = (p32 >> 3) & 1, h = (p32 & 7) + 8 * (p32 >> 4);
      int lane = n + 16 * g;
      size_t idx = (((size_t)(b * 16 + cblk)) * 128 + (p >> 5)) * 512 + lane * 16 + h;
      vBf[idx] = hv;
    }
  }
}

// ---------------------------------------------------------------------------
// Kernel B: energy (WMMA) + fused softmax over a 16-row strip held in LDS
// ---------------------------------------------------------------------------
__global__ void energy_softmax(const _Float16* __restrict__ qAf,
                               const _Float16* __restrict__ kBf,
                               float* __restrict__ attOut) {
  extern __shared__ float smem[];          // [16][LDSTRIDE] + red[256] + rowstat[16]
  float* red = smem + 16 * LDSTRIDE;
  float* rowstat = red + 256;
  int b = blockIdx.y;
  int rowblk = blockIdx.x;
  int t = threadIdx.x;                     // 256 threads, 8 waves
  int lane = t & 31;
  int wave = t >> 5;

  const v16h* qbase = (const v16h*)(qAf + ((size_t)(b * NB_ + rowblk)) * 2 * 512);
  v16h a0 = qbase[lane];                   // K = 0..31
  v16h a1 = qbase[32 + lane];              // K = 32..63
  int lr = lane >> 4, ln = lane & 15;

  for (int ct = 0; ct < 32; ++ct) {
    int colblk = wave * 32 + ct;
    const v16h* kbase = (const v16h*)(kBf + ((size_t)(b * NB_ + colblk)) * 2 * 512);
    v16h b0 = kbase[lane];
    v16h b1 = kbase[32 + lane];
    v8f acc = {};
    acc = wmma_f16(a0, b0, acc);
    acc = wmma_f16(a1, b1, acc);
    int col = colblk * 16 + ln;
    #pragma unroll
    for (int r = 0; r < 8; ++r)
      smem[(r + 8 * lr) * LDSTRIDE + col] = acc[r];
  }
  __syncthreads();

  int row = t & 15, seg = t >> 4;
  float* rp = smem + row * LDSTRIDE + seg * 256;
  float m = -3.0e38f;
  for (int k = 0; k < 256; ++k) m = fmaxf(m, rp[k]);
  red[t] = m;
  __syncthreads();
  if (t < 16) {
    float mm = red[t];
    for (int s2 = 1; s2 < 16; ++s2) mm = fmaxf(mm, red[t + 16 * s2]);
    rowstat[t] = mm;
  }
  __syncthreads();
  m = rowstat[row];
  float sum = 0.f;
  for (int k = 0; k < 256; ++k) { float e = __expf(rp[k] - m); rp[k] = e; sum += e; }
  red[t] = sum;
  __syncthreads();
  if (t < 16) {
    float ss = 0.f;
    for (int s2 = 0; s2 < 16; ++s2) ss += red[t + 16 * s2];
    rowstat[t] = 1.0f / ss;
  }
  __syncthreads();

  float* out = attOut + ((size_t)(b * N_ + rowblk * 16)) * N_;
  for (int iter = 0; iter < 64; ++iter) {   // coalesced float4 writes
    int idx = iter * 1024 + t * 4;
    int r2 = idx >> 12, col = idx & 4095;
    float inv = rowstat[r2];
    float4 v;
    v.x = smem[r2 * LDSTRIDE + col + 0] * inv;
    v.y = smem[r2 * LDSTRIDE + col + 1] * inv;
    v.z = smem[r2 * LDSTRIDE + col + 2] * inv;
    v.w = smem[r2 * LDSTRIDE + col + 3] * inv;
    *(float4*)(out + (size_t)r2 * N_ + col) = v;
  }
}

// ---------------------------------------------------------------------------
// Kernel C: out^T = att @ v^T (WMMA), epilogue gamma*acc + x
// ---------------------------------------------------------------------------
__global__ void attn_v(const float* __restrict__ att,
                       const _Float16* __restrict__ vBf,
                       const float* __restrict__ x,
                       const float* __restrict__ gamma,
                       float* __restrict__ out) {
  int b = blockIdx.y;
  int rowblk = blockIdx.x;
  int t = threadIdx.x;        // 128 threads, 4 waves
  int lane = t & 31;
  int wave = t >> 5;          // c-tiles [4*wave, 4*wave+4)
  int m = lane & 15, g = lane >> 4;

  v8f acc[4] = {v8f{}, v8f{}, v8f{}, v8f{}};
  const float* arow = att + ((size_t)(b * N_ + rowblk * 16 + m)) * N_;
  const size_t cstride = (size_t)128 * 512;

  for (int jc = 0; jc < 128; ++jc) {
    int j0 = jc * 32 + 8 * g;
    f32x4 r0a = *(const f32x4*)(arow + j0);
    f32x4 r0b = *(const f32x4*)(arow + j0 + 4);
    f32x4 r1a = *(const f32x4*)(arow + j0 + 16);
    f32x4 r1b = *(const f32x4*)(arow + j0 + 20);
    v16h a;
    a[0]  = (_Float16)r0a[0]; a[1]  = (_Float16)r0a[1];
    a[2]  = (_Float16)r0a[2]; a[3]  = (_Float16)r0a[3];
    a[4]  = (_Float16)r0b[0]; a[5]  = (_Float16)r0b[1];
    a[6]  = (_Float16)r0b[2]; a[7]  = (_Float16)r0b[3];
    a[8]  = (_Float16)r1a[0]; a[9]  = (_Float16)r1a[1];
    a[10] = (_Float16)r1a[2]; a[11] = (_Float16)r1a[3];
    a[12] = (_Float16)r1b[0]; a[13] = (_Float16)r1b[1];
    a[14] = (_Float16)r1b[2]; a[15] = (_Float16)r1b[3];

    size_t base = (((size_t)(b * 16 + wave * 4)) * 128 + jc) * 512;
    #pragma unroll
    for (int tt = 0; tt < 4; ++tt) {
      const v16h* vbp = (const v16h*)(vBf + base + (size_t)tt * cstride);
      acc[tt] = wmma_f16(a, vbp[lane], acc[tt]);
    }
  }

  float gm = gamma[0];
  #pragma unroll
  for (int tt = 0; tt < 4; ++tt) {
    int c = (wave * 4 + tt) * 16 + m;
    size_t off = ((size_t)(b * C_ + c)) * N_ + rowblk * 16 + 8 * g;
    #pragma unroll
    for (int r = 0; r < 8; ++r)
      out[off + r] = gm * acc[tt][r] + x[off + r];
  }
}

// ---------------------------------------------------------------------------
extern "C" void kernel_launch(void* const* d_in, const int* in_sizes, int n_in,
                              void* d_out, int out_size, void* d_ws, size_t ws_size,
                              hipStream_t stream) {
  const float* x   = (const float*)d_in[0];
  const float* qw  = (const float*)d_in[1];
  const float* qb  = (const float*)d_in[2];
  const float* kw  = (const float*)d_in[3];
  const float* kb  = (const float*)d_in[4];
  const float* vw  = (const float*)d_in[5];
  const float* vb  = (const float*)d_in[6];
  const float* c1w = (const float*)d_in[7];
  const float* c1b = (const float*)d_in[8];
  const float* c2w = (const float*)d_in[9];
  const float* c2b = (const float*)d_in[10];
  const float* c3w = (const float*)d_in[11];
  const float* c3b = (const float*)d_in[12];
  const float* c4w = (const float*)d_in[13];
  const float* c4b = (const float*)d_in[14];
  const float* gamma = (const float*)d_in[15];

  char* ws = (char*)d_ws;
  float* qcW = (float*)ws;                       // 64*256*4 = 65536 B
  float* qcB = (float*)(ws + 65536);             // 256 B
  float* kcW = (float*)(ws + 65792);             // 65536 B
  float* kcB = (float*)(ws + 131328);            // 256 B
  _Float16* qAf = (_Float16*)(ws + 131584);                 // 2 MB
  _Float16* kBf = (_Float16*)(ws + 131584 + 2097152);       // 2 MB
  _Float16* vBf = (_Float16*)(ws + 131584 + 2 * 2097152);   // 8 MB

  float* outMain = (float*)d_out;
  float* attOut  = outMain + (size_t)B_ * C_ * N_;   // attention after out

  combine_weights<<<1, 256, 0, stream>>>(qw, qb, kw, kb, c1w, c1b, c2w, c2b,
                                         c3w, c3b, c4w, c4b, qcW, qcB, kcW, kcB);

  dim3 gA(64, B_);
  proj_scatter<<<gA, 256, 0, stream>>>(x, qcW, qcB, kcW, kcB, vw, vb, qAf, kBf, vBf);

  dim3 gB(NB_, B_);
  size_t lds = (size_t)(16 * LDSTRIDE + 256 + 16) * sizeof(float);  // ~263 KB
  energy_softmax<<<gB, 256, lds, stream>>>(qAf, kBf, attOut);

  dim3 gC(NB_, B_);
  attn_v<<<gC, 128, 0, stream>>>(attOut, vBf, x, gamma, outMain);
}
